// HawkOptimized_73023033967375
// MI455X (gfx1250) — matrix operand order repaired and checked by reading, not compiled, not run on hardware
//
#include <hip/hip_runtime.h>
#include <math.h>

typedef __bf16 bf16_t;
typedef bf16_t v16bf __attribute__((ext_vector_type(16)));
typedef bf16_t v8bf  __attribute__((ext_vector_type(8)));
typedef float  v8f   __attribute__((ext_vector_type(8)));
typedef float  v4f   __attribute__((ext_vector_type(4)));
typedef int    v4i   __attribute__((ext_vector_type(4)));

#define AS1 __attribute__((address_space(1)))
#define AS3 __attribute__((address_space(3)))

// CDNA5 async global->LDS path (ASYNCcnt-tracked), with sync fallback.
#if defined(__has_builtin)
#  if __has_builtin(__builtin_amdgcn_global_load_async_to_lds_b128)
#    define HAVE_ASYNC_LDS 1
#  endif
#endif
#ifndef HAVE_ASYNC_LDS
#  define HAVE_ASYNC_LDS 0
#endif

namespace {

constexpr int Bn = 2048;
constexpr int Hn = 4096;
constexpr int THREADS = 256;      // 8 waves of 32 (wave32)
constexpr int KSTEP = 32;
constexpr int ROWE = 40;          // LDS row pitch in elements (80B: 20 banks, conflict-free)

union ABFrag { v16bf v; v8bf h[2]; };

struct EpiArgs {
    const float* conv_state;   // (B,H,3)
    const float* conv_w;       // (H,4)
    const float* conv_b;       // (H)
    const float* rglru_state;  // (B,H)
    const float* a_param;      // (H)
    const float* gate_f32;     // ws (B,H)
    const float* xconv_f32;    // ws (B,H)
    float* out;                // d_out (B,H)
    float* new_conv_state;     // d_out+B*H (B,H,3)
    float* new_rglru_state;    // d_out+4*B*H (B,H)
    float* gate_out;           // ws (B,H)
    float* xconv_out;          // ws (B,H)
    bf16_t* xconv_bf;          // ws (B,H)
    bf16_t* y_bf;              // ws (B,H)
};

__device__ __forceinline__ void async_b128(const bf16_t* g, bf16_t* l)
{
#if HAVE_ASYNC_LDS
    __builtin_amdgcn_global_load_async_to_lds_b128((AS1 v4i*)g, (AS3 v4i*)l, 0, 0);
#else
    *(v8bf*)l = *(const v8bf*)g;   // fallback: vmem load + ds_store
#endif
}

__device__ __forceinline__ void wait_async()
{
#if HAVE_ASYNC_LDS
#  if __has_builtin(__builtin_amdgcn_s_wait_asynccnt)
    __builtin_amdgcn_s_wait_asynccnt(0);
#  else
    asm volatile("s_wait_asynccnt 0" ::: "memory");
#  endif
#endif
}

__global__ __launch_bounds__(THREADS)
void cvt_f32_to_bf16(const float* __restrict__ in, bf16_t* __restrict__ out, long n8)
{
    long i = (long)blockIdx.x * blockDim.x + threadIdx.x;
    if (i >= n8) return;
    const v4f* p = (const v4f*)in;
    v4f a = p[2 * i];
    v4f b = p[2 * i + 1];
    v8bf o;
    o[0] = (bf16_t)a[0]; o[1] = (bf16_t)a[1]; o[2] = (bf16_t)a[2]; o[3] = (bf16_t)a[3];
    o[4] = (bf16_t)b[0]; o[5] = (bf16_t)b[1]; o[6] = (bf16_t)b[2]; o[7] = (bf16_t)b[3];
    ((v8bf*)out)[i] = o;
}

__device__ __forceinline__ float sigmoidf_fast(float x) {
    return 1.0f / (1.0f + __expf(-x));
}

// Fused elementwise epilogues operating on a (low, high) accumulator pair that
// holds columns (h) and (h+H) of the 2H-wide GEMM output.
template<int EPI>
__device__ __forceinline__ void epilogue_pair(const v8f& accLow, const v8f& accHigh,
                                              int h, int rBase, const EpiArgs& ea)
{
    if constexpr (EPI == 1) {
        // accLow: gate logits -> exact GELU; accHigh: x_recur -> causal conv.
        const float* cw = ea.conv_w + (size_t)h * 4;
        const float cw0 = cw[0], cw1 = cw[1], cw2 = cw[2], cw3 = cw[3];
        const float cb = ea.conv_b[h];
#pragma unroll
        for (int e = 0; e < 8; ++e) {
            const int b = rBase + e;
            const size_t off = (size_t)b * Hn + h;
            const float g = accLow[e];
            ea.gate_out[off] = 0.5f * g * (1.0f + erff(g * 0.70710678118654752f));
            const float xr = accHigh[e];
            const float* cs = ea.conv_state + off * 3;
            const float c0 = cs[0], c1 = cs[1], c2 = cs[2];
            const float xc = c0 * cw0 + c1 * cw1 + c2 * cw2 + xr * cw3 + cb;
            float* ncs = ea.new_conv_state + off * 3;
            ncs[0] = c1; ncs[1] = c2; ncs[2] = xr;
            ea.xconv_out[off] = xc;
            ea.xconv_bf[off]  = (bf16_t)xc;
        }
    } else { // EPI == 2, RG-LRU
        const float log_a = __logf(ea.a_param[h]);
#pragma unroll
        for (int e = 0; e < 8; ++e) {
            const int b = rBase + e;
            const size_t off = (size_t)b * Hn + h;
            const float it = sigmoidf_fast(accLow[e]);
            const float rt = sigmoidf_fast(accHigh[e]);
            const float at = __expf(8.0f * rt * log_a);
            const float mult = sqrtf(fmaxf(1.0f - at * at, 0.0f));
            const float xc = ea.xconv_f32[off];
            const float ns = ea.rglru_state[off] * at + mult * (it * xc);
            ea.new_rglru_state[off] = ns;
            ea.y_bf[off] = (bf16_t)(ea.gate_f32[off] * ns);
        }
    }
}

// LDS-tiled GEMM: block tile 128(M) x (64 low + 64 paired) cols x 32(K),
// 8 waves = 4(M) x 2(N); each wave: 2 M-subtiles x 4 N-subtiles (8 wmma/K-step).
// Global->LDS via async-to-LDS engine, double buffered, overlapped with WMMA.
template<int EPI>
__global__ __launch_bounds__(THREADS)
void wmma_gemm_lds(const bf16_t* __restrict__ A, const bf16_t* __restrict__ W,
                   int Kdim, int Mblocks, int nBlockStride, int pairOffRows, EpiArgs ea)
{
    __shared__ bf16_t sA[2][128 * ROWE];  // 2 x 10KB
    __shared__ bf16_t sB[2][128 * ROWE];  // 2 x 10KB

    const int tid   = threadIdx.x;
    const int lane  = tid & 31;
    const int wv    = tid >> 5;
    const int waveM = wv >> 1;         // 0..3
    const int waveN = wv & 1;          // 0..1
    const int row   = lane & 15;
    const int ksel  = lane >> 4;

    const int bm       = blockIdx.x % Mblocks;
    const int bn       = blockIdx.x / Mblocks;
    const int mBase    = bm * 128;
    const int lowBase  = bn * nBlockStride;
    const int highBase = lowBase + pairOffRows;

    // Per-thread staging map: thread t fills row r = t/2, 16-element chunk (t&1).
    const int r  = tid >> 1;                 // 0..127
    const int ce = (tid & 1) * 16;           // element offset in the 32-wide K slab
    const bf16_t* gArow = A + (size_t)(mBase + r) * Kdim + ce;
    const bf16_t* gBrow = (r < 64)
        ? (W + (size_t)(lowBase + r) * Kdim + ce)
        : (W + (size_t)(highBase + (r - 64)) * Kdim + ce);
    const int ldsOff = r * ROWE + ce;

    v8f acc[2][4] = {};

    const int nKB = Kdim / KSTEP;

    // Prologue: fill buffer 0.
    async_b128(gArow,     &sA[0][ldsOff]);
    async_b128(gArow + 8, &sA[0][ldsOff + 8]);
    async_b128(gBrow,     &sB[0][ldsOff]);
    async_b128(gBrow + 8, &sB[0][ldsOff + 8]);

    for (int kb = 0; kb < nKB; ++kb) {
        wait_async();
        __syncthreads();
        if (kb + 1 < nKB) {
            const int k = (kb + 1) * KSTEP;
            const int b = (kb + 1) & 1;
            async_b128(gArow + k,     &sA[b][ldsOff]);
            async_b128(gArow + k + 8, &sA[b][ldsOff + 8]);
            async_b128(gBrow + k,     &sB[b][ldsOff]);
            async_b128(gBrow + k + 8, &sB[b][ldsOff + 8]);
        }

        const bf16_t* aBuf = sA[kb & 1];
        const bf16_t* bBuf = sB[kb & 1];

        // A fragments (16-bit 16x32 layout): ksel*8 and +16 element chunks.
        ABFrag af[2];
#pragma unroll
        for (int ms = 0; ms < 2; ++ms) {
            const int base = (waveM * 32 + ms * 16 + row) * ROWE + ksel * 8;
            af[ms].h[0] = *(const v8bf*)(aBuf + base);
            af[ms].h[1] = *(const v8bf*)(aBuf + base + 16);
        }
        // B fragments: lane = column n; same 16-bit layout as A: ksel*8 and +16 element chunks.
        ABFrag bf4[4];
#pragma unroll
        for (int j = 0; j < 4; ++j) {
            const int trow = (j < 2) ? (waveN * 32 + j * 16)
                                     : (64 + waveN * 32 + (j - 2) * 16);
            const int base = (trow + row) * ROWE + ksel * 8;
            bf4[j].h[0] = *(const v8bf*)(bBuf + base);
            bf4[j].h[1] = *(const v8bf*)(bBuf + base + 16);
        }
#pragma unroll
        for (int ms = 0; ms < 2; ++ms)
#pragma unroll
            for (int j = 0; j < 4; ++j)
                acc[ms][j] = __builtin_amdgcn_wmma_f32_16x16x32_bf16(
                    false, af[ms].v, false, bf4[j].v, (short)0, acc[ms][j], false, false);
    }

    // Epilogues. C/D layout: element e -> row rBase+e, col base + (lane&15).
#pragma unroll
    for (int ms = 0; ms < 2; ++ms) {
        const int rBase = mBase + waveM * 32 + ms * 16 + ksel * 8;
        if constexpr (EPI == 3) {
#pragma unroll
            for (int j = 0; j < 4; ++j) {
                const int col = ((j < 2) ? (lowBase + waveN * 32 + j * 16)
                                         : (highBase + waveN * 32 + (j - 2) * 16)) + row;
#pragma unroll
                for (int e = 0; e < 8; ++e)
                    ea.out[(size_t)(rBase + e) * Hn + col] = acc[ms][j][e];
            }
        } else {
#pragma unroll
            for (int j = 0; j < 2; ++j) {
                const int h = lowBase + waveN * 32 + j * 16 + row;
                epilogue_pair<EPI>(acc[ms][j], acc[ms][j + 2], h, rBase, ea);
            }
        }
    }
}

} // namespace

extern "C" void kernel_launch(void* const* d_in, const int* in_sizes, int n_in,
                              void* d_out, int out_size, void* d_ws, size_t ws_size,
                              hipStream_t stream)
{
    const float* x          = (const float*)d_in[0];
    const float* conv_state = (const float*)d_in[1];
    const float* rglru      = (const float*)d_in[2];
    const float* w_fused    = (const float*)d_in[3];
    const float* conv_w     = (const float*)d_in[4];
    const float* conv_b     = (const float*)d_in[5];
    const float* w_gates    = (const float*)d_in[6];
    const float* a_param    = (const float*)d_in[7];
    const float* w_out      = (const float*)d_in[8];

    float* out   = (float*)d_out;
    float* newCS = out + (size_t)Bn * Hn;        // (B,H,3)
    float* newRS = out + (size_t)Bn * Hn * 4;    // (B,H)

    char* ws = (char*)d_ws;
    constexpr size_t SZ_XBF = (size_t)Bn * Hn * 2;          // 16 MB
    constexpr size_t SZ_WF  = (size_t)2 * Hn * Hn * 2;      // 64 MB
    constexpr size_t SZ_WG  = SZ_WF;                        // 64 MB
    constexpr size_t SZ_WO  = (size_t)Hn * Hn * 2;          // 32 MB
    constexpr size_t SZ_F32 = (size_t)Bn * Hn * 4;          // 32 MB
    bf16_t* xBf   = (bf16_t*)(ws);
    bf16_t* wfBf  = (bf16_t*)(ws + SZ_XBF);
    bf16_t* wgBf  = (bf16_t*)(ws + SZ_XBF + SZ_WF);
    bf16_t* woBf  = (bf16_t*)(ws + SZ_XBF + SZ_WF + SZ_WG);
    float*  gateF = (float*) (ws + SZ_XBF + SZ_WF + SZ_WG + SZ_WO);
    float*  xcF   = (float*) (ws + SZ_XBF + SZ_WF + SZ_WG + SZ_WO + SZ_F32);
    bf16_t* xcBf  = (bf16_t*)(ws + SZ_XBF + SZ_WF + SZ_WG + SZ_WO + 2 * SZ_F32);
    bf16_t* yBf   = (bf16_t*)(ws + SZ_XBF + SZ_WF + SZ_WG + SZ_WO + 2 * SZ_F32 + SZ_XBF);

    {
        const long n8x  = (long)Bn * Hn / 8;
        const long n8wf = (long)2 * Hn * Hn / 8;
        const long n8wo = (long)Hn * Hn / 8;
        cvt_f32_to_bf16<<<(int)(n8x  / THREADS), THREADS, 0, stream>>>(x,       xBf,  n8x);
        cvt_f32_to_bf16<<<(int)(n8wf / THREADS), THREADS, 0, stream>>>(w_fused, wfBf, n8wf);
        cvt_f32_to_bf16<<<(int)(n8wf / THREADS), THREADS, 0, stream>>>(w_gates, wgBf, n8wf);
        cvt_f32_to_bf16<<<(int)(n8wo / THREADS), THREADS, 0, stream>>>(w_out,   woBf, n8wo);
    }

    EpiArgs ea;
    ea.conv_state      = conv_state;
    ea.conv_w          = conv_w;
    ea.conv_b          = conv_b;
    ea.rglru_state     = rglru;
    ea.a_param         = a_param;
    ea.gate_f32        = gateF;
    ea.xconv_f32       = xcF;
    ea.out             = out;
    ea.new_conv_state  = newCS;
    ea.new_rglru_state = newRS;
    ea.gate_out        = gateF;
    ea.xconv_out       = xcF;
    ea.xconv_bf        = xcBf;
    ea.y_bf            = yBf;

    constexpr int Mblocks = Bn / 128;                 // 16
    constexpr int blocks12 = Mblocks * (Hn / 64);     // 1024: N covers 64 low + 64 paired(+H)
    constexpr int blocks3  = Mblocks * (Hn / 128);    // 512:  N covers 128 contiguous

    // GEMM1: fused = x @ w_fused^T; pairs (n, n+H): GELU-gate + causal-conv epilogue
    wmma_gemm_lds<1><<<blocks12, THREADS, 0, stream>>>(xBf, wfBf, Hn, Mblocks, 64, Hn, ea);
    // GEMM2: gates = x_conv @ w_gates^T; pairs (i_t, r_t): RG-LRU epilogue
    wmma_gemm_lds<2><<<blocks12, THREADS, 0, stream>>>(xcBf, wgBf, Hn, Mblocks, 64, Hn, ea);
    // GEMM3: out = (gate * state) @ w_out^T; plain store over 128 contiguous cols
    wmma_gemm_lds<3><<<blocks3, THREADS, 0, stream>>>(yBf, woBf, Hn, Mblocks, 128, 64, ea);
}
